// MSDABlock_25967372272267
// MI455X (gfx1250) — matrix operand order repaired and checked by reading, not compiled, not run on hardware
//
#include <hip/hip_runtime.h>

// ---------------------------------------------------------------------------
// MSDA block for gfx1250 (MI455X). bf16 WMMA GEMMs + coalesced deformable
// sampling. Sizes fixed by the harness: B=16, H=W=64, N=4096, C=256, M=8,
// hd=32, K=4.
//
// GEMM structure: 256-thread block (8 waves) owns 32 rows x ALL cout columns.
// A-tile staged once in LDS (activations read from HBM exactly once per
// GEMM); the compiler hoists all A fragments into registers across column
// tiles. Per column tile, ALL 8 B fragments are loaded up front (one clause)
// so one load-latency wait covers 16 back-to-back WMMAs.
// ---------------------------------------------------------------------------

#define BB   16
#define HH   64
#define WW   64
#define NN   4096          // H*W
#define CC   256
#define MM   8
#define HD   32
#define KP   4
#define BN   (BB * NN)     // 65536
#define OACOLS 96          // M*K*2 (=64) offsets + M*K (=32) attn logits
#define ROWT 2             // row tiles (of 16) per block -> 32 rows
#define KT   (CC / 32)     // 8 k-steps

typedef unsigned short u16;
typedef __attribute__((ext_vector_type(16))) __bf16 v16bf;
typedef __attribute__((ext_vector_type(8)))  float  v8f;

union frag_t { uint4 u[2]; v16bf v; };

__device__ __forceinline__ u16 f32_to_bf16(float f) {
    union { float f; unsigned u; } v; v.f = f;
    unsigned r = v.u + 0x7FFFu + ((v.u >> 16) & 1u);   // round-to-nearest-even
    return (u16)(r >> 16);
}

// ---------------------------------------------------------------------------
// Weight cast + transpose: src (Kdim x Ncols, f32 row-major) ->
// dst (Ncols x Kdim, bf16 row-major), so B-fragments load contiguously.
// ---------------------------------------------------------------------------
__global__ void cast_transpose_kernel(const float* __restrict__ src,
                                      u16* __restrict__ dst,
                                      int Kdim, int Ncols) {
    int i = blockIdx.x * blockDim.x + threadIdx.x;
    if (i >= Kdim * Ncols) return;
    int k = i / Ncols;
    int c = i - k * Ncols;
    dst[(size_t)c * Kdim + k] = f32_to_bf16(src[i]);
}

// ---------------------------------------------------------------------------
// LayerNorm: one 256-thread block per row, output bf16.
// ---------------------------------------------------------------------------
__global__ __launch_bounds__(CC) void ln_kernel(const float* __restrict__ x,
                                                const float* __restrict__ g,
                                                const float* __restrict__ b,
                                                u16* __restrict__ xn) {
    __shared__ float red[CC];
    const int row = blockIdx.x;
    const int c   = threadIdx.x;
    float v = x[(size_t)row * CC + c];
    red[c] = v;
    __syncthreads();
    for (int s = CC / 2; s > 0; s >>= 1) {
        if (c < s) red[c] += red[c + s];
        __syncthreads();
    }
    float mu = red[0] * (1.0f / CC);
    __syncthreads();
    float d = v - mu;
    red[c] = d * d;
    __syncthreads();
    for (int s = CC / 2; s > 0; s >>= 1) {
        if (c < s) red[c] += red[c + s];
        __syncthreads();
    }
    float var = red[0] * (1.0f / CC);
    float r = rsqrtf(var + 1e-5f);
    xn[(size_t)row * CC + c] = f32_to_bf16(d * r * g[c] + b[c]);
}

// ---------------------------------------------------------------------------
// WMMA GEMM: out[BN x cout] = A[BN x 256](bf16) @ Wt^T (Wt stored cout x 256
// bf16) + bias.  v_wmma_f32_16x16x32_bf16; two row tiles share each B
// fragment; all B fragments for a column tile loaded in one clause.
// Epilogue modes:
//   0: out16[row*cout+col] = bf16(acc+bias)                       (q)
//   1: out32 laid out (B, M, N, hd) for coalesced sampling        (v)
//   2: out32[row*cout+col] = acc                                  (off/attn raw)
//   3: out32[row*cout+col] = resid + acc + bias                   (final)
// ---------------------------------------------------------------------------
#define LDA_PAD (CC + 8)   // 264 halfwords -> 528B row stride, 16B aligned
__global__ __launch_bounds__(256) void wmma_gemm_kernel(
    const u16* __restrict__ A, const u16* __restrict__ Wt,
    const float* __restrict__ bias, int cout, int mode,
    u16* __restrict__ out16, float* __restrict__ out32,
    const float* __restrict__ resid) {
    __shared__ u16 smA[16 * ROWT][LDA_PAD];

    const int row0 = blockIdx.x * (16 * ROWT);

    // cooperative A-tile load: 32 rows x 256 bf16 = 16 KB, uint4 chunks
    for (int i = threadIdx.x; i < 16 * ROWT * (CC / 8); i += blockDim.x) {
        int r  = i >> 5;           // CC/8 == 32 chunks per row
        int ch = i & 31;
        uint4 d = *(const uint4*)(A + (size_t)(row0 + r) * CC + ch * 8);
        *(uint4*)(&smA[r][ch * 8]) = d;
    }
    __syncthreads();

    const int wave  = threadIdx.x >> 5;
    const int lane  = threadIdx.x & 31;
    const int lrow  = lane & 15;
    const int lhalf = lane >> 4;
    const int cout_tiles = cout >> 4;

    for (int ci = wave; ci < cout_tiles; ci += 8) {
        const u16* wcol = Wt + (size_t)(ci * 16 + lrow) * CC;  // one B column

        // load ALL B fragments for this column tile up front (one clause)
        frag_t bf[KT];
        #pragma unroll
        for (int kt = 0; kt < KT; ++kt) {
            const int kbb = kt * 32 + lhalf * 16;   // 16 contiguous K values
            bf[kt].u[0] = *(const uint4*)(wcol + kbb);
            bf[kt].u[1] = *(const uint4*)(wcol + kbb + 8);
        }

        v8f acc[ROWT];
        #pragma unroll
        for (int rt = 0; rt < ROWT; ++rt) acc[rt] = (v8f){};

        #pragma unroll
        for (int kt = 0; kt < KT; ++kt) {
            const int kb = kt * 32 + lhalf * 16;
            #pragma unroll
            for (int rt = 0; rt < ROWT; ++rt) {
                frag_t af;   // elems 0..15 -> K=[kb,kb+16)
                af.u[0] = *(const uint4*)(&smA[rt * 16 + lrow][kb]);
                af.u[1] = *(const uint4*)(&smA[rt * 16 + lrow][kb + 8]);
                acc[rt] = __builtin_amdgcn_wmma_f32_16x16x32_bf16(
                    false, af.v, false, bf[kt].v, (short)0, acc[rt], false, false);
            }
        }

        const int col  = ci * 16 + lrow;       // N position = lane&15
        const float bc = bias ? bias[col] : 0.0f;

        #pragma unroll
        for (int rt = 0; rt < ROWT; ++rt) {
            #pragma unroll
            for (int j = 0; j < 8; ++j) {
                const int mrow = j + lhalf * 8;   // C/D layout: VGPR j, lane half
                const int row  = row0 + rt * 16 + mrow;
                float val = acc[rt][j] + bc;
                if (mode == 0) {
                    out16[(size_t)row * cout + col] = f32_to_bf16(val);
                } else if (mode == 1) {
                    int b_ = row >> 12, n_ = row & (NN - 1);
                    int m_ = col >> 5, d_ = col & (HD - 1);
                    out32[((size_t)(b_ * MM + m_) * NN + n_) * HD + d_] = val;
                } else if (mode == 2) {
                    out32[(size_t)row * cout + col] = val;
                } else {
                    size_t idx = (size_t)row * cout + col;
                    out32[idx] = resid[idx] + val;
                }
            }
        }
    }
}

// ---------------------------------------------------------------------------
// Deformable sampling: one wave per (b, n, m); lane = channel d in [0,32).
// Reads raw q@[Wo|Wa] logits, applies bias+tanh / softmax in-register, then
// 4-point bilinear gather from v in (B, M, N, hd) layout (coalesced 128B per
// pixel across the wave).  Writes bf16 mid (B, N, C).
// ---------------------------------------------------------------------------
__global__ __launch_bounds__(256) void sample_kernel(
    const float* __restrict__ oa,        // [BN][96]
    const float* __restrict__ bo,        // [64]
    const float* __restrict__ ba,        // [32]
    const float* __restrict__ vfeat,     // [B][M][N][32]
    u16* __restrict__ mid) {             // [BN][256]
    const int gwave = blockIdx.x * 8 + (threadIdx.x >> 5);
    const int lane  = threadIdx.x & 31;
    const int m     = gwave & (MM - 1);
    const int rown  = gwave >> 3;               // b*N + n
    const int b_    = rown >> 12;
    const int n_    = rown & (NN - 1);

    const float* oarow = oa + (size_t)rown * OACOLS;
    const float px = ((float)(n_ & (WW - 1)) + 0.5f) * (1.0f / WW);
    const float py = ((float)(n_ >> 6)       + 0.5f) * (1.0f / HH);

    // softmax over K=4 attention logits
    float lg[KP];
    float mx = -1e30f;
    #pragma unroll
    for (int k = 0; k < KP; ++k) {
        lg[k] = oarow[64 + m * KP + k] + ba[m * KP + k];
        mx = fmaxf(mx, lg[k]);
    }
    float se = 0.0f;
    #pragma unroll
    for (int k = 0; k < KP; ++k) { lg[k] = __expf(lg[k] - mx); se += lg[k]; }
    const float inv_se = 1.0f / se;

    const float* vbase = vfeat + (size_t)(b_ * MM + m) * NN * HD;
    float acc = 0.0f;

    #pragma unroll
    for (int k = 0; k < KP; ++k) {
        float ox = tanhf(oarow[m * 8 + k * 2 + 0] + bo[m * 8 + k * 2 + 0]) * 0.5f;
        float oy = tanhf(oarow[m * 8 + k * 2 + 1] + bo[m * 8 + k * 2 + 1]) * 0.5f;
        float lx = 2.0f * (px + ox) - 1.0f;
        float ly = 2.0f * (py + oy) - 1.0f;
        // grid_sample, align_corners=False
        float ix = ((lx + 1.0f) * WW - 1.0f) * 0.5f;
        float iy = ((ly + 1.0f) * HH - 1.0f) * 0.5f;
        float x0f = floorf(ix), y0f = floorf(iy);
        float fx = ix - x0f, fy = iy - y0f;
        int x0 = (int)x0f, y0 = (int)y0f;
        float s = 0.0f;
        #pragma unroll
        for (int dy = 0; dy < 2; ++dy) {
            #pragma unroll
            for (int dx = 0; dx < 2; ++dx) {
                int xi = x0 + dx, yi = y0 + dy;
                if (xi >= 0 && xi < WW && yi >= 0 && yi < HH) {
                    float w = (dx ? fx : 1.0f - fx) * (dy ? fy : 1.0f - fy);
                    s += w * vbase[((size_t)(yi * WW + xi)) * HD + lane];
                }
            }
        }
        acc += lg[k] * inv_se * s;
    }
    mid[(size_t)rown * CC + m * HD + lane] = f32_to_bf16(acc);
}

// ---------------------------------------------------------------------------
// Host-side orchestration (graph-capture safe: launches only).
// ---------------------------------------------------------------------------
extern "C" void kernel_launch(void* const* d_in, const int* in_sizes, int n_in,
                              void* d_out, int out_size, void* d_ws, size_t ws_size,
                              hipStream_t stream) {
    const float* x     = (const float*)d_in[0];
    const float* gamma = (const float*)d_in[1];
    const float* beta  = (const float*)d_in[2];
    const float* Wq    = (const float*)d_in[3];
    const float* bq    = (const float*)d_in[4];
    const float* Wv    = (const float*)d_in[5];
    const float* bv    = (const float*)d_in[6];
    const float* Wo    = (const float*)d_in[7];
    const float* bo    = (const float*)d_in[8];
    const float* Wa    = (const float*)d_in[9];
    const float* ba    = (const float*)d_in[10];
    const float* Wout  = (const float*)d_in[11];
    const float* bout  = (const float*)d_in[12];
    float* out = (float*)d_out;

    // workspace carve-up (bytes)
    char* ws = (char*)d_ws;
    size_t off = 0;
    u16*   xn    = (u16*)(ws + off); off += (size_t)BN * CC * 2;        // 33.5 MB
    u16*   WqT   = (u16*)(ws + off); off += (size_t)CC * CC * 2;        // 128 KB
    u16*   WvT   = (u16*)(ws + off); off += (size_t)CC * CC * 2;
    u16*   WoaT  = (u16*)(ws + off); off += (size_t)OACOLS * CC * 2;    // 48 KB
    u16*   WoutT = (u16*)(ws + off); off += (size_t)CC * CC * 2;
    u16*   qb    = (u16*)(ws + off); off += (size_t)BN * CC * 2;        // 33.5 MB
    float* vfeat = (float*)(ws + off); off += (size_t)BN * CC * 4;      // 67 MB
    float* oa    = (float*)(ws + off); off += (size_t)BN * OACOLS * 4;  // 25 MB
    u16*   mid   = (u16*)(ws + off); off += (size_t)BN * CC * 2;        // 33.5 MB
    (void)off; (void)ws_size; (void)in_sizes; (void)n_in; (void)out_size;

    // 1) weight casts (bf16, transposed to cout x 256)
    cast_transpose_kernel<<<(CC * CC + 255) / 256, 256, 0, stream>>>(Wq, WqT, CC, CC);
    cast_transpose_kernel<<<(CC * CC + 255) / 256, 256, 0, stream>>>(Wv, WvT, CC, CC);
    cast_transpose_kernel<<<(CC * 64 + 255) / 256, 256, 0, stream>>>(Wo, WoaT, CC, 64);
    cast_transpose_kernel<<<(CC * 32 + 255) / 256, 256, 0, stream>>>(Wa, WoaT + 64 * CC, CC, 32);
    cast_transpose_kernel<<<(CC * CC + 255) / 256, 256, 0, stream>>>(Wout, WoutT, CC, CC);

    // 2) layernorm -> bf16
    ln_kernel<<<BN, CC, 0, stream>>>(x, gamma, beta, xn);

    const int gemm_blocks = BN / (16 * ROWT);   // 2048

    // 3) q = xn @ Wq + bq  (bf16 out)
    wmma_gemm_kernel<<<gemm_blocks, 256, 0, stream>>>(
        xn, WqT, bq, CC, 0, qb, nullptr, nullptr);

    // 4) v = xn @ Wv + bv, scattered to (B, M, N, hd)
    wmma_gemm_kernel<<<gemm_blocks, 256, 0, stream>>>(
        xn, WvT, bv, CC, 1, nullptr, vfeat, nullptr);

    // 5) raw off/attn logits: q @ [Wo|Wa]  (biases applied in sampler)
    wmma_gemm_kernel<<<gemm_blocks, 256, 0, stream>>>(
        qb, WoaT, nullptr, OACOLS, 2, nullptr, oa, nullptr);

    // 6) deformable bilinear sampling + softmax-weighted sum -> mid (bf16)
    sample_kernel<<<BN, 256, 0, stream>>>(oa, bo, ba, vfeat, mid);

    // 7) out = x + mid @ Wout + bout
    wmma_gemm_kernel<<<gemm_blocks, 256, 0, stream>>>(
        mid, WoutT, bout, CC, 3, nullptr, out, x);
}
